// SimpleMoE_91010357002866
// MI455X (gfx1250) — matrix operand order repaired and checked by reading, not compiled, not run on hardware
//
#include <hip/hip_runtime.h>

// SimpleMoE on MI455X (gfx1250, wave32, WMMA).
// B=2, S=1024 -> 2048 tokens, D=1024, F=2048, E=8.
// Route top-1, gather tokens per expert, run only the routed expert's MLP
// with v_wmma_f32_16x16x32_bf16 (fp32 operands converted to bf16 on the fly;
// bf16 keeps fp32's exponent range).
//
// Roofline: 17.2 GFLOP routed GEMM vs ~145 MB mandatory traffic (weights fit
// in the 192 MB L2) -> ~6-7 us at either roof. Round-2 disasm showed each
// k-step serialized issue-loads -> wait -> cvt -> wmma with only 1 wave/SIMD.
// This round: 8 waves/block (2 waves/SIMD despite 192 KB LDS) + explicit
// register pipelining of the B fragment (issue next k-step's raw fp32 loads,
// run the two WMMAs of the current step, convert afterwards) so the
// s_wait_loadcnt overlaps XDL execution instead of preceding it.

#define D_DIM 1024
#define F_DIM 2048
#define E_NUM 8
#define NTOK  2048
#define TM    32      // token rows per tile = 2 x WMMA M

typedef __attribute__((ext_vector_type(16))) __bf16 bf16x16;
typedef __attribute__((ext_vector_type(8)))  __bf16 bf16x8;
typedef __attribute__((ext_vector_type(8)))  float  f32x8;

// ---------------------------------------------------------------------------
__global__ void moe_init_kernel(int* __restrict__ counts) {
    if (threadIdx.x < E_NUM) counts[threadIdx.x] = 0;
}

// ---------------------------------------------------------------------------
// Router: one wave32 per token; shfl_xor reduction (wave32), strict-> argmax
// (first-max tie-break, matches jnp.argmax), atomic append to expert list.
__global__ __launch_bounds__(256) void moe_router_kernel(
        const float* __restrict__ x,  const float* __restrict__ Wr,
        const float* __restrict__ br, int* __restrict__ counts,
        int* __restrict__ list) {
    const int lane = threadIdx.x & 31;
    const int wave = threadIdx.x >> 5;
    const int tok  = blockIdx.x * 8 + wave;
    const float* xr = x + (size_t)tok * D_DIM;

    float s[E_NUM];
#pragma unroll
    for (int e = 0; e < E_NUM; ++e) s[e] = 0.0f;

    for (int j = 0; j < D_DIM / 32; ++j) {
        const int d = lane + j * 32;
        const float xv = xr[d];
        const float* wr = Wr + (size_t)d * E_NUM;   // 8 contiguous floats
#pragma unroll
        for (int e = 0; e < E_NUM; ++e) s[e] += xv * wr[e];
    }
#pragma unroll
    for (int off = 16; off > 0; off >>= 1) {
#pragma unroll
        for (int e = 0; e < E_NUM; ++e) s[e] += __shfl_xor(s[e], off, 32);
    }
    if (lane == 0) {
        int   best = 0;
        float bv   = s[0] + br[0];
#pragma unroll
        for (int e = 1; e < E_NUM; ++e) {
            const float v = s[e] + br[e];
            if (v > bv) { bv = v; best = e; }   // strict >: first max wins
        }
        const int pos = atomicAdd(&counts[best], 1);
        list[best * NTOK + pos] = tok;
    }
}

// ---------------------------------------------------------------------------
// Expert MLP: one block = one expert e + one tile of up to 32 routed tokens
// (two 16-row WMMA tiles sharing every B fragment). 8 waves split N.
__global__ __launch_bounds__(256) void moe_expert_kernel(
        const float* __restrict__ x,
        const float* __restrict__ W1, const float* __restrict__ b1,
        const float* __restrict__ W2, const float* __restrict__ b2,
        const int* __restrict__ counts, const int* __restrict__ list,
        float* __restrict__ out) {
    __shared__ __bf16 Xs[TM * D_DIM];   //  64 KB: token rows of x (bf16)
    __shared__ __bf16 Hs[TM * F_DIM];   // 128 KB: relu hidden (bf16)
    __shared__ int    toks[TM];

    const int e    = blockIdx.x >> 6;      // 64 tiles per expert
    const int tile = blockIdx.x & 63;
    const int cnt  = counts[e];
    const int base = tile * TM;
    if (base >= cnt) return;               // block-uniform early exit
    const int nval = min(TM, cnt - base);

    const int tid  = threadIdx.x;
    const int lane = tid & 31;
    const int wave = tid >> 5;             // 0..7
    const int mrow = lane & 15;            // WMMA M / N index within lane half
    const int half = lane >> 4;            // lane group 0 / 1

    if (tid < TM) {
        const int idx = (tid < nval) ? tid : (nval - 1);   // clamp padding rows
        toks[tid] = list[e * NTOK + base + idx];
    }
    __syncthreads();

    // Stage x tile -> LDS bf16 (32 rows x 1024 cols, float4 vectorized).
    for (int i = tid; i < TM * (D_DIM / 4); i += 256) {
        const int row = i >> 8;            // 256 float4 per row
        const int c4  = (i & 255) * 4;
        const float4 v = *(const float4*)(x + (size_t)toks[row] * D_DIM + c4);
        __bf16* dst = &Xs[row * D_DIM + c4];
        dst[0] = (__bf16)v.x; dst[1] = (__bf16)v.y;
        dst[2] = (__bf16)v.z; dst[3] = (__bf16)v.w;
    }
    __syncthreads();

    // ---------------- GEMM1: Hs[32,F] = relu(Xs[32,D] * W1[e][D,F] + b1) ----
    const float* W1e = W1 + (size_t)e * D_DIM * F_DIM;
    for (int nt = wave; nt < F_DIM / 16; nt += 8) {
        const int col = nt * 16 + mrow;    // this lane's N column
        f32x8 acc0 = {};                   // token rows 0..15
        f32x8 acc1 = {};                   // token rows 16..31
        // Prologue: load + convert B fragment for k0 = 0.
        bf16x16 bcur;
#pragma unroll
        for (int i = 0; i < 16; ++i)
            bcur[i] = (__bf16)W1e[(size_t)(half * 16 + i) * F_DIM + col];
        for (int k0 = 0; k0 < D_DIM; k0 += 32) {
            // Issue next k-step's raw fp32 B loads first (no wait yet).
            float nxt[16];
            const bool more = (k0 + 32) < D_DIM;
            if (more) {
                const int kbn = k0 + 32 + half * 16;
#pragma unroll
                for (int i = 0; i < 16; ++i)
                    nxt[i] = W1e[(size_t)(kbn + i) * F_DIM + col];
                __builtin_prefetch(&W1e[(size_t)(kbn + 32) * F_DIM + col], 0, 0);
            }
            // A fragments: contiguous k = 16*half + s (matches B fragment).
            const __bf16* a0p = &Xs[mrow * D_DIM + k0 + half * 16];
            const __bf16* a1p = a0p + 16 * D_DIM;
            const bf16x16 a0 = __builtin_shufflevector(
                *(const bf16x8*)a0p, *(const bf16x8*)(a0p + 8),
                0,1,2,3,4,5,6,7,8,9,10,11,12,13,14,15);
            const bf16x16 a1 = __builtin_shufflevector(
                *(const bf16x8*)a1p, *(const bf16x8*)(a1p + 8),
                0,1,2,3,4,5,6,7,8,9,10,11,12,13,14,15);
            acc0 = __builtin_amdgcn_wmma_f32_16x16x32_bf16(
                false, a0, false, bcur, (short)0, acc0, false, false);
            acc1 = __builtin_amdgcn_wmma_f32_16x16x32_bf16(
                false, a1, false, bcur, (short)0, acc1, false, false);
            // Convert arrived data after the WMMAs were issued.
            if (more) {
                bf16x16 bn;
#pragma unroll
                for (int i = 0; i < 16; ++i) bn[i] = (__bf16)nxt[i];
                bcur = bn;
            }
        }
        const float bias = b1[e * F_DIM + col];
#pragma unroll
        for (int r = 0; r < 8; ++r) {
            float v0 = acc0[r] + bias;
            float v1 = acc1[r] + bias;
            v0 = v0 > 0.0f ? v0 : 0.0f;                          // ReLU
            v1 = v1 > 0.0f ? v1 : 0.0f;
            Hs[(r + 8 * half) * F_DIM + col]      = (__bf16)v0;
            Hs[(16 + r + 8 * half) * F_DIM + col] = (__bf16)v1;
        }
    }
    __syncthreads();

    // ---------------- GEMM2: out = Hs[32,F] * W2[e][F,D] + b2 ---------------
    const float* W2e = W2 + (size_t)e * F_DIM * D_DIM;
    for (int nt = wave; nt < D_DIM / 16; nt += 8) {
        const int col = nt * 16 + mrow;
        f32x8 acc0 = {};
        f32x8 acc1 = {};
        bf16x16 bcur;
#pragma unroll
        for (int i = 0; i < 16; ++i)
            bcur[i] = (__bf16)W2e[(size_t)(half * 16 + i) * D_DIM + col];
        for (int k0 = 0; k0 < F_DIM; k0 += 32) {
            float nxt[16];
            const bool more = (k0 + 32) < F_DIM;
            if (more) {
                const int kbn = k0 + 32 + half * 16;
#pragma unroll
                for (int i = 0; i < 16; ++i)
                    nxt[i] = W2e[(size_t)(kbn + i) * D_DIM + col];
                __builtin_prefetch(&W2e[(size_t)(kbn + 32) * D_DIM + col], 0, 0);
            }
            const __bf16* a0p = &Hs[mrow * F_DIM + k0 + half * 16];
            const __bf16* a1p = a0p + 16 * F_DIM;
            const bf16x16 a0 = __builtin_shufflevector(
                *(const bf16x8*)a0p, *(const bf16x8*)(a0p + 8),
                0,1,2,3,4,5,6,7,8,9,10,11,12,13,14,15);
            const bf16x16 a1 = __builtin_shufflevector(
                *(const bf16x8*)a1p, *(const bf16x8*)(a1p + 8),
                0,1,2,3,4,5,6,7,8,9,10,11,12,13,14,15);
            acc0 = __builtin_amdgcn_wmma_f32_16x16x32_bf16(
                false, a0, false, bcur, (short)0, acc0, false, false);
            acc1 = __builtin_amdgcn_wmma_f32_16x16x32_bf16(
                false, a1, false, bcur, (short)0, acc1, false, false);
            if (more) {
                bf16x16 bn;
#pragma unroll
                for (int i = 0; i < 16; ++i) bn[i] = (__bf16)nxt[i];
                bcur = bn;
            }
        }
        const float bias = b2[e * D_DIM + col];
#pragma unroll
        for (int r = 0; r < 8; ++r) {
            const int row0 = r + 8 * half;
            const int row1 = 16 + row0;
            if (row0 < nval)
                out[(size_t)toks[row0] * D_DIM + col] = acc0[r] + bias;
            if (row1 < nval)
                out[(size_t)toks[row1] * D_DIM + col] = acc1[r] + bias;
        }
    }
}

// ---------------------------------------------------------------------------
extern "C" void kernel_launch(void* const* d_in, const int* in_sizes, int n_in,
                              void* d_out, int out_size, void* d_ws, size_t ws_size,
                              hipStream_t stream) {
    const float* x  = (const float*)d_in[0];   // [B,S,D]
    const float* W1 = (const float*)d_in[1];   // [E,D,F]
    const float* b1 = (const float*)d_in[2];   // [E,F]
    const float* W2 = (const float*)d_in[3];   // [E,F,D]
    const float* b2 = (const float*)d_in[4];   // [E,D]
    const float* Wr = (const float*)d_in[5];   // [D,E]
    const float* br = (const float*)d_in[6];   // [E]
    float* out = (float*)d_out;                // [B,S,D]

    int* counts = (int*)d_ws;                  // 8 ints
    int* list   = counts + 8;                  // [E, NTOK] token ids

    moe_init_kernel<<<1, 32, 0, stream>>>(counts);
    moe_router_kernel<<<NTOK / 8, 256, 0, stream>>>(x, Wr, br, counts, list);
    moe_expert_kernel<<<E_NUM * 64, 256, 0, stream>>>(
        x, W1, b1, W2, b2, counts, list, out);
}
